// ParticleEmbedding_9526237463164
// MI455X (gfx1250) — matrix operand order repaired and checked by reading, not compiled, not run on hardware
//
#include <hip/hip_runtime.h>

// =============================================================================
// ParticleEmbedding on MI455X (gfx1250, wave32, WMMA).
//
// Algebraic fusion (exact): per-pair score hidden u = hr @ Wf + q_i + k_j + c0,
//   Wf = Wv2*J0*Ws1, q = (emb@M0)@(J1*Ws1), k = (emb@M1)@(J1*Ws1),
//   c0 = bv2@(J0*Ws1)+bs1;  s = relu(u)@Ws2 + bs2.
// LayerNorm stats of the 2->64 value layer are closed-form quadratics of the two
// invariants (9 precomputed moments). Attention output collapses to
//   out_i = (S_i@Wv2+bv2)@J0 + (mi + Mw_i)@J1,  S_i = sum_j a_ij hr_ij.
//
// WMMA operands are staged in LDS pre-swizzled into fragment order so each
// fragment loads as 2x ds_load_b128 per lane. GEMM epilogue variants are
// compile-time templated (no per-element uniform branches).
//
// Workspace requirement: ~5.1 MB.
//
// Input order assumption: d_in[0]=coords, d_in[1]=particle_info, then JAX
// tree_leaves(params) with sorted dict keys:
//   per attn: J0,J1,M0,M1, score{W1,W2,b1,b2}, value{W1,W2,b1,b2,beta,gamma}
//   per nl  : W1,W2,b1,b2,beta,gamma
//   params  : blocks[0](attn,nl), blocks[1](attn,nl), final(attn), info_W, info_b
// =============================================================================

#define B_ 2
#define N_ 512
#define D_ 64

typedef __bf16 v16bf __attribute__((ext_vector_type(16)));
typedef unsigned short v16us __attribute__((ext_vector_type(16)));
typedef float v8f __attribute__((ext_vector_type(8)));

__device__ __forceinline__ unsigned short f32_to_bf16(float f) {
  unsigned int x = __float_as_uint(f);
  unsigned int r = x + 0x7FFFu + ((x >> 16) & 1u);
  return (unsigned short)(r >> 16);
}

__device__ __forceinline__ v8f wmma_bf16(v16us a, v16us b, v8f c) {
  return __builtin_amdgcn_wmma_f32_16x16x32_bf16(
      false, __builtin_bit_cast(v16bf, a),
      false, __builtin_bit_cast(v16bf, b),
      (short)0, c, false, false);
}

// A-fragment K map (ISA 7.12.2, 16-bit A 16x32): element h, lane-half lh -> K.
__device__ __forceinline__ int kmapA(int h, int lh) {
  int r = h >> 1;
  int base = (r < 4) ? (2 * r) : (2 * r + 8);
  return base + (lh ? 8 : 0) + (h & 1);
}
// Inverse: K (0..31) -> (lane-half lh, element h).
__device__ __forceinline__ void amapInv(int k, int& lh, int& h) {
  lh = (k >> 3) & 1;
  int kk = k - 8 * lh; // in {0..7} U {16..23}
  int r = (kk < 8) ? (kk >> 1) : ((kk >> 1) - 4);
  h = 2 * r + (kk & 1);
}

// -------------------- fused per-layer constant block (floats) ----------------
#define LAYER_WF 0      // 64x64  Wv2*J0*Ws1   (row-major [k][n])
#define LAYER_WJS 4096  // 64x64  J1*Ws1
#define LAYER_C0 8192   // 64     bv2@(J0*Ws1)+bs1
#define LAYER_WS2 8256  // 64
#define LAYER_MISC 8320 // 10: bs2, a0,a1,ab, m00,m11,m01,m0b,m1b,mbb
#define LAYER_WV10 8336 // 64  Wv1 row0
#define LAYER_WV11 8400 // 64  Wv1 row1
#define LAYER_BV1 8464  // 64
#define LAYER_GAM 8528  // 64
#define LAYER_BET 8592  // 64
#define LAYER_SIZE 8704

// -------------------- workspace layout (float offsets) -----------------------
#define W_LAYERS 0u
#define W_EMB 26112u
#define W_MI (W_EMB + 65536u)
#define W_MJ (W_MI + 65536u)
#define W_Q (W_MJ + 65536u)
#define W_K (W_Q + 65536u)
#define W_SC (W_K + 65536u)   // B*N*N scores/probs
#define W_S (W_SC + 524288u)  // per-row Sum_j a*hr
#define W_U (W_S + 65536u)    // per-row mi + Sum_j a*mj  (or a*(mi+mj))
#define W_P (W_U + 65536u)
#define W_A (W_P + 65536u)
#define W_H (W_A + 65536u)
#define W_HR (W_H + 65536u)
#define W_GST (W_HR + 65536u) // gmax[B], gsum[B]
#define W_SG (W_GST + 4u)     // B*64
#define W_MM (W_SG + 128u)    // B*64

struct AttnRaw {
  const float *J0, *J1, *M0, *M1, *sW1, *sW2, *sb1, *sb2;
  const float *vW1, *vW2, *vb1, *vb2, *vbeta, *vgamma;
};
struct FuseArgs { AttnRaw a[3]; float* Lbuf; };

struct LnStats { float a0, a1, ab, m00, m11, m01, m0b, m1b, mbb; };

__device__ __forceinline__ void pair_inv(float cix, float ciy, float ciz,
                                         float x, float y, float z,
                                         float& inv0, float& inv1) {
  inv0 = cix * x + ciy * y + ciz * z;
  float crx = ciy * z - ciz * y;
  float cry = ciz * x - cix * z;
  float crz = cix * y - ciy * x;
  float sq = crx * crx + cry * cry + crz * crz;
  inv1 = sq > 0.f ? sqrtf(sq) : 0.f; // gradient-safe norm semantics
}

__device__ __forceinline__ void ln_mv(const LnStats& s, float inv0, float inv1,
                                      float& mean, float& rstd) {
  mean = inv0 * s.a0 + inv1 * s.a1 + s.ab;
  float eh2 = inv0 * inv0 * s.m00 + inv1 * inv1 * s.m11 + s.mbb +
              2.f * (inv0 * inv1 * s.m01 + inv0 * s.m0b + inv1 * s.m1b);
  float var = eh2 - mean * mean;
  rstd = rsqrtf(var + 1e-3f);
}

// Swizzled B-weight store: element (k,n) of a 64x64 matrix -> fragment slot.
// Layout: v16us index = (ntile*2 + frag)*32 + lane, lane = ((ksub>>3)&1)*16 + (n&15).
__device__ __forceinline__ int bswz_index(int k, int n) {
  int frag = k >> 5, kr = k & 31;
  int lane = (((kr >> 3) & 1) << 4) | (n & 15);
  int ntile = n >> 4;
  return ((((ntile << 1) | frag) << 5) | lane) * 16 + ((kr & 7) | ((kr >> 4) << 3));
}

// ===================== kernel: fuse weights (one block per attn layer) =======
__global__ __launch_bounds__(64) void fuse_weights(FuseArgs fa) {
  __shared__ float T[64 * 64];
  int l = blockIdx.x, t = threadIdx.x;
  AttnRaw p = fa.a[l];
  float* L = fa.Lbuf + (size_t)l * LAYER_SIZE;
  for (int k = 0; k < 64; ++k) { // T = J0 @ Ws1
    float acc = 0.f;
    for (int q = 0; q < 64; ++q) acc += p.J0[k * 64 + q] * p.sW1[q * 64 + t];
    T[k * 64 + t] = acc;
  }
  __syncthreads();
  for (int f = 0; f < 64; ++f) { // Wf = Wv2 @ T
    float acc = 0.f;
    for (int q = 0; q < 64; ++q) acc += p.vW2[f * 64 + q] * T[q * 64 + t];
    L[LAYER_WF + f * 64 + t] = acc;
  }
  {
    float acc = 0.f;
    for (int q = 0; q < 64; ++q) acc += p.vb2[q] * T[q * 64 + t];
    L[LAYER_C0 + t] = acc + p.sb1[t];
  }
  __syncthreads();
  for (int k = 0; k < 64; ++k) { // WJS = J1 @ Ws1
    float acc = 0.f;
    for (int q = 0; q < 64; ++q) acc += p.J1[k * 64 + q] * p.sW1[q * 64 + t];
    L[LAYER_WJS + k * 64 + t] = acc;
  }
  L[LAYER_WS2 + t] = p.sW2[t]; // (64,1) column
  L[LAYER_WV10 + t] = p.vW1[t];
  L[LAYER_WV11 + t] = p.vW1[64 + t];
  L[LAYER_BV1 + t] = p.vb1[t];
  L[LAYER_GAM + t] = p.vgamma[t];
  L[LAYER_BET + t] = p.vbeta[t];
  if (t == 0) {
    float s0 = 0, s1 = 0, sb = 0, q00 = 0, q11 = 0, q01 = 0, q0b = 0, q1b = 0, qbb = 0;
    for (int f = 0; f < 64; ++f) {
      float w0 = p.vW1[f], w1 = p.vW1[64 + f], b = p.vb1[f];
      s0 += w0; s1 += w1; sb += b;
      q00 += w0 * w0; q11 += w1 * w1; q01 += w0 * w1;
      q0b += w0 * b; q1b += w1 * b; qbb += b * b;
    }
    const float iv = 1.f / 64.f;
    L[LAYER_MISC + 0] = p.sb2[0];
    L[LAYER_MISC + 1] = s0 * iv; L[LAYER_MISC + 2] = s1 * iv; L[LAYER_MISC + 3] = sb * iv;
    L[LAYER_MISC + 4] = q00 * iv; L[LAYER_MISC + 5] = q11 * iv; L[LAYER_MISC + 6] = q01 * iv;
    L[LAYER_MISC + 7] = q0b * iv; L[LAYER_MISC + 8] = q1b * iv; L[LAYER_MISC + 9] = qbb * iv;
  }
}

// ===================== kernel: generic 1024x64 @ 64x64 WMMA GEMM =============
// Y = X @ W (+bias) (+Z).  128 rows/block, 8 waves, 16 rows/wave.
// Both operands staged in LDS in fragment-swizzled bf16 order; epilogue
// variants are compile-time (no uniform branches around element loads).
struct GemmArgs { const float* X; const float* Wt; const float* bias; const float* Z; float* Y; };

template <bool HAS_BIAS, bool HAS_Z>
__global__ __launch_bounds__(256) void gemm64(GemmArgs g) {
  __shared__ v16us Xv[8 * 2 * 32]; // [wave][frag][lane] : 16 KB
  __shared__ v16us Wv[4 * 2 * 32]; // [ntile][frag][lane]:  8 KB
  unsigned short* Xswz = (unsigned short*)Xv;
  unsigned short* Wswz = (unsigned short*)Wv;
  int tid = threadIdx.x;
  int row0 = blockIdx.x * 128;
  for (int idx = tid; idx < 128 * 64; idx += 256) {
    int rowl = idx >> 6, k = idx & 63;
    int wave = rowl >> 4, m = rowl & 15;
    int frag = k >> 5, kr = k & 31;
    int lh, h;
    amapInv(kr, lh, h);
    int lane = lh * 16 + m;
    Xswz[((((wave << 1) | frag) << 5) | lane) * 16 + h] =
        f32_to_bf16(g.X[(size_t)row0 * 64 + idx]);
  }
  for (int idx = tid; idx < 64 * 64; idx += 256) {
    int k = idx >> 6, n = idx & 63;
    Wswz[bswz_index(k, n)] = f32_to_bf16(g.Wt[idx]);
  }
  __syncthreads();
  int wid = tid >> 5, lane = tid & 31, lanelo = lane & 15, lh = lane >> 4;
  v16us a0u = Xv[(wid * 2 + 0) * 32 + lane];
  v16us a1u = Xv[(wid * 2 + 1) * 32 + lane];
#pragma unroll
  for (int n = 0; n < 4; ++n) {
    int N0 = lanelo + 16 * n;
    v16us b0u = Wv[(n * 2 + 0) * 32 + lane];
    v16us b1u = Wv[(n * 2 + 1) * 32 + lane];
    float binit = HAS_BIAS ? g.bias[N0] : 0.f; // one load, invariant over rows
    v8f acc;
#pragma unroll
    for (int r = 0; r < 8; ++r) {
      float v = binit;
      if (HAS_Z) {
        int grow = row0 + wid * 16 + r + 8 * lh;
        v += g.Z[(size_t)grow * 64 + N0];
      }
      acc[r] = v;
    }
    acc = wmma_bf16(a0u, b0u, acc);
    acc = wmma_bf16(a1u, b1u, acc);
#pragma unroll
    for (int r = 0; r < 8; ++r) {
      int grow = row0 + wid * 16 + r + 8 * lh;
      g.Y[(size_t)grow * 64 + N0] = acc[r];
    }
  }
}

// ===================== kernel: pair scores (WMMA), one block per (b,i) =======
struct ScoreArgs { const float* coords; const float* kArr; const float* qArr;
                   const float* L; float* scores; };

__global__ __launch_bounds__(256) void pass_scores(ScoreArgs A) {
  __shared__ v16us WfV[4 * 2 * 32]; // fragment-swizzled Wf (8 KB)
  __shared__ float cjs[N_ * 3];
  __shared__ float qc[64], ws2[64], wv10[64], wv11[64], bv1s[64], gam[64], bet[64];
  unsigned short* Wswz = (unsigned short*)WfV;
  int row = blockIdx.x; // b*N + i
  int b = row >> 9, i = row & 511;
  int tid = threadIdx.x;
  for (int idx = tid; idx < 64 * 64; idx += 256) {
    int k = idx >> 6, n = idx & 63;
    Wswz[bswz_index(k, n)] = f32_to_bf16(A.L[LAYER_WF + idx]);
  }
  for (int idx = tid; idx < N_ * 3; idx += 256) cjs[idx] = A.coords[(size_t)b * N_ * 3 + idx];
  if (tid < 64) {
    qc[tid] = A.qArr[(size_t)row * 64 + tid] + A.L[LAYER_C0 + tid];
    ws2[tid] = A.L[LAYER_WS2 + tid];
    wv10[tid] = A.L[LAYER_WV10 + tid];
    wv11[tid] = A.L[LAYER_WV11 + tid];
    bv1s[tid] = A.L[LAYER_BV1 + tid];
    gam[tid] = A.L[LAYER_GAM + tid];
    bet[tid] = A.L[LAYER_BET + tid];
  }
  __syncthreads();
  LnStats st = {A.L[LAYER_MISC + 1], A.L[LAYER_MISC + 2], A.L[LAYER_MISC + 3],
                A.L[LAYER_MISC + 4], A.L[LAYER_MISC + 5], A.L[LAYER_MISC + 6],
                A.L[LAYER_MISC + 7], A.L[LAYER_MISC + 8], A.L[LAYER_MISC + 9]};
  float bs2v = A.L[LAYER_MISC + 0];
  float cix = cjs[i * 3 + 0], ciy = cjs[i * 3 + 1], ciz = cjs[i * 3 + 2];
  bool mask_i = (cix != 0.f) || (ciy != 0.f) || (ciz != 0.f);
  int wid = tid >> 5, lane = tid & 31, lanelo = lane & 15, lh = lane >> 4;

  for (int jt = wid; jt < N_ / 16; jt += 8) {
    int j0 = jt * 16;
    int j = j0 + lanelo;
    float inv0, inv1;
    pair_inv(cix, ciy, ciz, cjs[j * 3], cjs[j * 3 + 1], cjs[j * 3 + 2], inv0, inv1);
    float mean, rstd;
    ln_mv(st, inv0, inv1, mean, rstd);
    v16us a0u, a1u; // hr in A-fragment layout (computed in-register)
#pragma unroll
    for (int h = 0; h < 16; ++h) {
      int kk = kmapA(h, lh);
      float h0 = inv0 * wv10[kk] + inv1 * wv11[kk] + bv1s[kk];
      a0u[h] = f32_to_bf16(fmaxf(0.f, (h0 - mean) * rstd * gam[kk] + bet[kk]));
      int k1 = kk + 32;
      float h1 = inv0 * wv10[k1] + inv1 * wv11[k1] + bv1s[k1];
      a1u[h] = f32_to_bf16(fmaxf(0.f, (h1 - mean) * rstd * gam[k1] + bet[k1]));
    }
    float spart[8];
#pragma unroll
    for (int r = 0; r < 8; ++r) spart[r] = 0.f;
#pragma unroll
    for (int n = 0; n < 4; ++n) {
      int N0 = lanelo + 16 * n;
      v16us b0u = WfV[(n * 2 + 0) * 32 + lane];
      v16us b1u = WfV[(n * 2 + 1) * 32 + lane];
      v8f acc;
#pragma unroll
      for (int r = 0; r < 8; ++r) {
        int jr = j0 + r + 8 * lh;
        acc[r] = qc[N0] + A.kArr[((size_t)b * N_ + jr) * 64 + N0];
      }
      acc = wmma_bf16(a0u, b0u, acc);
      acc = wmma_bf16(a1u, b1u, acc);
#pragma unroll
      for (int r = 0; r < 8; ++r) spart[r] += fmaxf(acc[r], 0.f) * ws2[N0];
    }
#pragma unroll
    for (int r = 0; r < 8; ++r) { // reduce over N within each 16-lane half
      spart[r] += __shfl_xor(spart[r], 8, 32);
      spart[r] += __shfl_xor(spart[r], 4, 32);
      spart[r] += __shfl_xor(spart[r], 2, 32);
      spart[r] += __shfl_xor(spart[r], 1, 32);
    }
    if (lanelo == 0) {
#pragma unroll
      for (int r = 0; r < 8; ++r) {
        int jr = j0 + r + 8 * lh;
        float x = cjs[jr * 3], y = cjs[jr * 3 + 1], z = cjs[jr * 3 + 2];
        bool mj = (x != 0.f) || (y != 0.f) || (z != 0.f);
        float sc = spart[r] + bs2v;
        if (!(mask_i && mj)) sc = -3.402823466e38f;
        A.scores[(size_t)row * N_ + jr] = sc;
      }
    }
  }
}

// ===================== kernel: per-row masked softmax (in place) =============
__global__ __launch_bounds__(256) void row_softmax(float* scores) {
  __shared__ float red[256];
  int row = blockIdx.x, tid = threadIdx.x;
  float* s = scores + (size_t)row * N_;
  float m = -3.402823466e38f;
  for (int j = tid; j < N_; j += 256) m = fmaxf(m, s[j]);
  red[tid] = m; __syncthreads();
  for (int k = 128; k > 0; k >>= 1) { if (tid < k) red[tid] = fmaxf(red[tid], red[tid + k]); __syncthreads(); }
  float gm = red[0]; __syncthreads();
  float sum = 0.f;
  for (int j = tid; j < N_; j += 256) { float e = expf(s[j] - gm); s[j] = e; sum += e; }
  red[tid] = sum; __syncthreads();
  for (int k = 128; k > 0; k >>= 1) { if (tid < k) red[tid] += red[tid + k]; __syncthreads(); }
  float inv = 1.f / red[0];
  for (int j = tid; j < N_; j += 256) s[j] *= inv;
}

// ===================== kernel: global softmax stats (one block per batch) ====
__global__ __launch_bounds__(256) void global_stats(const float* scores, float* gstats) {
  __shared__ float red[256];
  int b = blockIdx.x, tid = threadIdx.x;
  const float* s = scores + (size_t)b * N_ * N_;
  float m = -3.402823466e38f;
  for (int idx = tid; idx < N_ * N_; idx += 256) m = fmaxf(m, s[idx]);
  red[tid] = m; __syncthreads();
  for (int k = 128; k > 0; k >>= 1) { if (tid < k) red[tid] = fmaxf(red[tid], red[tid + k]); __syncthreads(); }
  float gm = red[0]; __syncthreads();
  float sum = 0.f;
  for (int idx = tid; idx < N_ * N_; idx += 256) sum += expf(s[idx] - gm);
  red[tid] = sum; __syncthreads();
  for (int k = 128; k > 0; k >>= 1) { if (tid < k) red[tid] += red[tid + k]; __syncthreads(); }
  if (tid == 0) { gstats[b] = gm; gstats[2 + b] = red[0]; }
}

// ===================== kernel: weighted accumulation pass ====================
// equiv mode: Sout_i = sum_j a*hr, Uout_i = mi + sum_j a*mj  (a = probs in-place)
// global mode: a = exp(s-gmax)/gsum, Uout_i = sum_j a*(mi+mj)
struct PassCArgs { const float* coords; const float* scores; const float* mi; const float* mj;
                   const float* L; const float* gstats; float* Sout; float* Uout; int global_mode; };

__global__ __launch_bounds__(256) void passC(PassCArgs A) {
  __shared__ float cjs[N_ * 3];
  __shared__ float wv10[64], wv11[64], bv1s[64], gam[64], bet[64];
  __shared__ float sh[4][64], uh[4][64];
  int row = blockIdx.x, b = row >> 9, i = row & 511, tid = threadIdx.x;
  for (int idx = tid; idx < N_ * 3; idx += 256) cjs[idx] = A.coords[(size_t)b * N_ * 3 + idx];
  if (tid < 64) {
    wv10[tid] = A.L[LAYER_WV10 + tid]; wv11[tid] = A.L[LAYER_WV11 + tid];
    bv1s[tid] = A.L[LAYER_BV1 + tid];
    gam[tid] = A.L[LAYER_GAM + tid]; bet[tid] = A.L[LAYER_BET + tid];
  }
  __syncthreads();
  LnStats st = {A.L[LAYER_MISC + 1], A.L[LAYER_MISC + 2], A.L[LAYER_MISC + 3],
                A.L[LAYER_MISC + 4], A.L[LAYER_MISC + 5], A.L[LAYER_MISC + 6],
                A.L[LAYER_MISC + 7], A.L[LAYER_MISC + 8], A.L[LAYER_MISC + 9]};
  float cix = cjs[i * 3], ciy = cjs[i * 3 + 1], ciz = cjs[i * 3 + 2];
  int f = tid & 63, jj = tid >> 6;
  float gm = 0.f, ginv = 1.f;
  if (A.global_mode) { gm = A.gstats[b]; ginv = 1.f / A.gstats[2 + b]; }
  float miF = A.mi[(size_t)row * 64 + f];
  float sAcc = 0.f, uAcc = 0.f;
  for (int j = jj; j < N_; j += 4) {
    float sc = A.scores[(size_t)row * N_ + j];
    float p = A.global_mode ? (expf(sc - gm) * ginv) : sc;
    float inv0, inv1;
    pair_inv(cix, ciy, ciz, cjs[j * 3], cjs[j * 3 + 1], cjs[j * 3 + 2], inv0, inv1);
    float mean, rstd;
    ln_mv(st, inv0, inv1, mean, rstd);
    float hv = inv0 * wv10[f] + inv1 * wv11[f] + bv1s[f];
    float hr = fmaxf(0.f, (hv - mean) * rstd * gam[f] + bet[f]);
    sAcc += p * hr;
    float mjF = A.mj[((size_t)b * N_ + j) * 64 + f];
    uAcc += A.global_mode ? p * (miF + mjF) : p * mjF;
  }
  sh[jj][f] = sAcc; uh[jj][f] = uAcc; __syncthreads();
  if (jj == 0) {
    float s = sh[0][f] + sh[1][f] + sh[2][f] + sh[3][f];
    float u = uh[0][f] + uh[1][f] + uh[2][f] + uh[3][f];
    A.Sout[(size_t)row * 64 + f] = s;
    A.Uout[(size_t)row * 64 + f] = A.global_mode ? u : (miF + u);
  }
}

// ===================== kernel: deterministic row reduction over i ============
__global__ __launch_bounds__(64) void reduce_rows(const float* Sb, const float* Mb,
                                                  float* Sg, float* MMg) {
  int b = blockIdx.x, f = threadIdx.x;
  float s = 0.f, m = 0.f;
  for (int i = 0; i < N_; ++i) {
    s += Sb[((size_t)b * N_ + i) * 64 + f];
    m += Mb[((size_t)b * N_ + i) * 64 + f];
  }
  Sg[b * 64 + f] = s;
  MMg[b * 64 + f] = m;
}

// ===================== kernel: LayerNorm + relu over rows of 64 ==============
__global__ __launch_bounds__(64) void ln_relu(const float* X, const float* gamma,
                                              const float* beta, float* Y) {
  __shared__ float red[64];
  int row = blockIdx.x, t = threadIdx.x;
  float x = X[(size_t)row * 64 + t];
  red[t] = x; __syncthreads();
  for (int k = 32; k > 0; k >>= 1) { if (t < k) red[t] += red[t + k]; __syncthreads(); }
  float mean = red[0] * (1.f / 64.f); __syncthreads();
  float d = x - mean;
  red[t] = d * d; __syncthreads();
  for (int k = 32; k > 0; k >>= 1) { if (t < k) red[t] += red[t + k]; __syncthreads(); }
  float var = red[0] * (1.f / 64.f);
  Y[(size_t)row * 64 + t] = fmaxf(0.f, d * rsqrtf(var + 1e-3f) * gamma[t] + beta[t]);
}

// ===================== kernel: info embedding (K=8, scalar) ==================
__global__ __launch_bounds__(256) void embed_k(const float* info, const float* W,
                                               const float* bias, float* emb) {
  int idx = blockIdx.x * 256 + threadIdx.x;
  if (idx >= B_ * N_ * 64) return;
  int r = idx >> 6, n = idx & 63;
  float acc = bias[n];
  for (int f = 0; f < 8; ++f) acc += info[r * 8 + f] * W[f * 64 + n];
  emb[idx] = acc;
}

// ===================== kernel: final invariant output ========================
struct FinishArgs { const float* Sg; const float* MM; const float* vW2; const float* vb2;
                    const float* J0; const float* J1; float* out; };

__global__ __launch_bounds__(64) void finish_k(FinishArgs a) {
  int b = blockIdx.x, n = threadIdx.x;
  const float* S = a.Sg + b * 64;
  const float* M = a.MM + b * 64;
  float acc = 0.f;
  for (int k = 0; k < 64; ++k) {
    float pk = a.vb2[k];
    for (int f = 0; f < 64; ++f) pk += S[f] * a.vW2[f * 64 + k];
    acc += pk * a.J0[k * 64 + n] + M[k] * a.J1[k * 64 + n];
  }
  a.out[b * 64 + n] = acc;
}

// =============================================================================
extern "C" void kernel_launch(void* const* d_in, const int* in_sizes, int n_in,
                              void* d_out, int out_size, void* d_ws, size_t ws_size,
                              hipStream_t stream) {
  (void)in_sizes; (void)n_in; (void)out_size; (void)ws_size;
  float* ws = (float*)d_ws;
  const float* coords = (const float*)d_in[0];
  const float* pinfo = (const float*)d_in[1];
  auto F = [&](int i) { return (const float*)d_in[i]; };

  // params leaves (JAX sorted-key order), bases: blocks[0].attn=2, nl=16,
  // blocks[1].attn=22, nl=36, final.attn=42, info_W=56, info_b=57
  AttnRaw attn[3];
  const int bases[3] = {2, 22, 42};
  for (int l = 0; l < 3; ++l) {
    int o = bases[l];
    attn[l] = {F(o + 0), F(o + 1), F(o + 2), F(o + 3),  // J0,J1,M0,M1
               F(o + 4), F(o + 5), F(o + 6), F(o + 7),  // score W1,W2,b1,b2
               F(o + 8), F(o + 9), F(o + 10), F(o + 11),// value W1,W2,b1,b2
               F(o + 12), F(o + 13)};                   // value beta,gamma
  }
  const float* nlW1[2] = {F(16), F(36)};
  const float* nlW2[2] = {F(17), F(37)};
  const float* nlb1[2] = {F(18), F(38)};
  const float* nlb2[2] = {F(19), F(39)};
  const float* nlbt[2] = {F(20), F(40)};
  const float* nlgm[2] = {F(21), F(41)};
  const float* infoW = F(56);
  const float* infoB = F(57);

  const int gblk = (B_ * N_) / 128;
  auto gemm = [&](const float* X, const float* Wt, const float* bias, const float* Z, float* Y) {
    GemmArgs g{X, Wt, bias, Z, Y};
    if (bias && Z)       gemm64<true, true><<<gblk, 256, 0, stream>>>(g);
    else if (bias)       gemm64<true, false><<<gblk, 256, 0, stream>>>(g);
    else if (Z)          gemm64<false, true><<<gblk, 256, 0, stream>>>(g);
    else                 gemm64<false, false><<<gblk, 256, 0, stream>>>(g);
  };

  FuseArgs fa; fa.a[0] = attn[0]; fa.a[1] = attn[1]; fa.a[2] = attn[2];
  fa.Lbuf = ws + W_LAYERS;
  fuse_weights<<<3, 64, 0, stream>>>(fa);
  embed_k<<<(B_ * N_ * 64 + 255) / 256, 256, 0, stream>>>(pinfo, infoW, infoB, ws + W_EMB);

  for (int l = 0; l < 2; ++l) {
    const float* L = ws + W_LAYERS + (size_t)l * LAYER_SIZE;
    gemm(ws + W_EMB, attn[l].M0, nullptr, nullptr, ws + W_MI);
    gemm(ws + W_EMB, attn[l].M1, nullptr, nullptr, ws + W_MJ);
    gemm(ws + W_MI, L + LAYER_WJS, nullptr, nullptr, ws + W_Q);
    gemm(ws + W_MJ, L + LAYER_WJS, nullptr, nullptr, ws + W_K);
    ScoreArgs sa{coords, ws + W_K, ws + W_Q, L, ws + W_SC};
    pass_scores<<<B_ * N_, 256, 0, stream>>>(sa);
    row_softmax<<<B_ * N_, 256, 0, stream>>>(ws + W_SC);
    PassCArgs pc{coords, ws + W_SC, ws + W_MI, ws + W_MJ, L, nullptr, ws + W_S, ws + W_U, 0};
    passC<<<B_ * N_, 256, 0, stream>>>(pc);
    gemm(ws + W_S, attn[l].vW2, attn[l].vb2, nullptr, ws + W_P); // P = S@Wv2+bv2
    gemm(ws + W_P, attn[l].J0, nullptr, nullptr, ws + W_A);      // A = P@J0
    gemm(ws + W_U, attn[l].J1, nullptr, ws + W_A, ws + W_A);     // A += U@J1
    gemm(ws + W_A, nlW1[l], nlb1[l], nullptr, ws + W_H);         // H = A@W1+b1
    ln_relu<<<B_ * N_, 64, 0, stream>>>(ws + W_H, nlgm[l], nlbt[l], ws + W_HR);
    gemm(ws + W_HR, nlW2[l], nlb2[l], ws + W_EMB, ws + W_EMB);   // emb = Hr@W2+b2+emb
  }

  // final attention (reduce=True, global softmax)
  const float* L2 = ws + W_LAYERS + 2 * (size_t)LAYER_SIZE;
  gemm(ws + W_EMB, attn[2].M0, nullptr, nullptr, ws + W_MI);
  gemm(ws + W_EMB, attn[2].M1, nullptr, nullptr, ws + W_MJ);
  gemm(ws + W_MI, L2 + LAYER_WJS, nullptr, nullptr, ws + W_Q);
  gemm(ws + W_MJ, L2 + LAYER_WJS, nullptr, nullptr, ws + W_K);
  ScoreArgs sa{coords, ws + W_K, ws + W_Q, L2, ws + W_SC};
  pass_scores<<<B_ * N_, 256, 0, stream>>>(sa);
  global_stats<<<B_, 256, 0, stream>>>(ws + W_SC, ws + W_GST);
  PassCArgs pc{coords, ws + W_SC, ws + W_MI, ws + W_MJ, L2, ws + W_GST, ws + W_S, ws + W_U, 1};
  passC<<<B_ * N_, 256, 0, stream>>>(pc);
  reduce_rows<<<B_, 64, 0, stream>>>(ws + W_S, ws + W_U, ws + W_SG, ws + W_MM);
  FinishArgs fin{ws + W_SG, ws + W_MM, attn[2].vW2, attn[2].vb2, attn[2].J0, attn[2].J1,
                 (float*)d_out};
  finish_k<<<B_, 64, 0, stream>>>(fin);
}
